// DeformableLayer_16844861735643
// MI455X (gfx1250) — compile-verified, matrix operand order checked by reading, not run on hardware
//
#include <hip/hip_runtime.h>
#include <hip/hip_bf16.h>

typedef __attribute__((ext_vector_type(16))) _Float16 v16h;
typedef __attribute__((ext_vector_type(8)))  float    v8f;

#define B_   32
#define C_   192
#define H_   64
#define W_   64
#define N_   4096
#define CR_  12
#define LN_EPS 1e-5f
#define HSTR 197   // 64-pixel tile row stride (197 mod 64 = 5, coprime -> conflict-free)

__device__ __forceinline__ float gelu_exact(float v) {
    return 0.5f * v * (1.0f + erff(v * 0.70710678118654752440f));
}

// ---------------- Kernel 1: channel means xc[b,c] = mean_{h,w} x ----------------
__global__ __launch_bounds__(256) void k_mean(const float* __restrict__ x,
                                              float* __restrict__ xc) {
    __shared__ float red[256];
    const int bc = blockIdx.x;                 // b*C + c
    const float* p = x + (size_t)bc * N_;
    float s = 0.f;
    for (int n = threadIdx.x; n < N_; n += 256) s += p[n];
    red[threadIdx.x] = s;
    __syncthreads();
    for (int w = 128; w > 0; w >>= 1) {
        if (threadIdx.x < w) red[threadIdx.x] += red[threadIdx.x + w];
        __syncthreads();
    }
    if (threadIdx.x == 0) xc[bc] = red[0] * (1.0f / (float)N_);
}

// ---------------- Kernel 2: channel attention a = sigmoid(gelu(xc W1^T+b1) W2^T + b2) ----
__global__ __launch_bounds__(C_) void k_attn(const float* __restrict__ xc,
                                             const float* __restrict__ w1,
                                             const float* __restrict__ b1,
                                             const float* __restrict__ w2,
                                             const float* __restrict__ b2,
                                             float* __restrict__ a) {
    __shared__ float sx[C_];
    __shared__ float st[CR_];
    const int b = blockIdx.x, t = threadIdx.x;
    sx[t] = xc[b * C_ + t];
    __syncthreads();
    if (t < CR_) {
        float s = b1[t];
        for (int c = 0; c < C_; ++c) s += w1[t * C_ + c] * sx[c];
        st[t] = gelu_exact(s);
    }
    __syncthreads();
    float s = b2[t];
    #pragma unroll
    for (int r = 0; r < CR_; ++r) s += w2[t * CR_ + r] * st[r];
    a[b * C_ + t] = 1.0f / (1.0f + expf(-s));
}

// ---------------- Kernel 3: rpe64[c,64,64] = bilinear resize of rpe_table[c,7,7] --------
__global__ __launch_bounds__(256) void k_rpe(const float* __restrict__ tab,
                                             float* __restrict__ rpe64) {
    const int c = blockIdx.x;
    const float* tc = tab + c * 49;
    for (int p = threadIdx.x; p < N_; p += 256) {
        const int i = p >> 6, j = p & 63;
        float sy = (i + 0.5f) * (7.0f / 64.0f) - 0.5f;
        float sx = (j + 0.5f) * (7.0f / 64.0f) - 0.5f;
        float fy = floorf(sy), fx = floorf(sx);
        float wy = sy - fy, wx = sx - fx;
        int y0 = (int)fy, x0 = (int)fx;
        int y0c = min(max(y0, 0), 6), y1c = min(max(y0 + 1, 0), 6);
        int x0c = min(max(x0, 0), 6), x1c = min(max(x0 + 1, 0), 6);
        float v00 = tc[y0c * 7 + x0c], v01 = tc[y0c * 7 + x1c];
        float v10 = tc[y1c * 7 + x0c], v11 = tc[y1c * 7 + x1c];
        rpe64[(size_t)c * N_ + p] =
            (v00 * (1.f - wx) + v01 * wx) * (1.f - wy) +
            (v10 * (1.f - wx) + v11 * wx) * wy;
    }
}

// ---------------- Kernel 4: depthwise 9x9 conv (+bias) * a  -> h ------------------------
__global__ __launch_bounds__(256) void k_conv(const float* __restrict__ x,
                                              const float* __restrict__ w,
                                              const float* __restrict__ bias,
                                              const float* __restrict__ a,
                                              float* __restrict__ h) {
    __shared__ float sIn[16 * 40];   // (8+8) x (32+8) input tile, zero-padded
    __shared__ float sW[81];
    const int blk  = blockIdx.x;
    const int tile = blk & 15;
    const int c    = (blk >> 4) % C_;
    const int b    = blk / (16 * C_);
    const int y0   = (tile >> 1) * 8;
    const int x0   = (tile & 1) * 32;
    const float* xp = x + ((size_t)(b * C_ + c)) * N_;
    for (int i = threadIdx.x; i < 640; i += 256) {
        int r = i / 40, cc = i - r * 40;
        int gy = y0 - 4 + r, gx = x0 - 4 + cc;
        float v = 0.f;
        if (gy >= 0 && gy < H_ && gx >= 0 && gx < W_) v = xp[gy * W_ + gx];
        sIn[i] = v;
    }
    if (threadIdx.x < 81) sW[threadIdx.x] = w[c * 81 + threadIdx.x];
    __syncthreads();
    const int oy = threadIdx.x >> 5, ox = threadIdx.x & 31;
    float acc = 0.f;
    #pragma unroll
    for (int ky = 0; ky < 9; ++ky)
        #pragma unroll
        for (int kx = 0; kx < 9; ++kx)
            acc = fmaf(sIn[(oy + ky) * 40 + ox + kx], sW[ky * 9 + kx], acc);
    float out = (acc + bias[c]) * a[b * C_ + c];
    h[((size_t)(b * C_ + c)) * N_ + (y0 + oy) * W_ + x0 + ox] = out;
}

// ---------------- Kernel 5: fused LN + GELU + WMMA projection + pos + dual grid-sample --
// Each block: one image row (64 pixels) x all 192 channels, 128 threads = 4 waves.
// Writes sampled output in place over h (its h slab was fully consumed into LDS first).
__global__ __launch_bounds__(128) void k_fused(const float* __restrict__ x,
                                               float* __restrict__ h,      // in: h, out: xo
                                               const float* __restrict__ ln_g,
                                               const float* __restrict__ ln_b,
                                               const float* __restrict__ w2, // conv2_w (3,C)
                                               const float* __restrict__ rpe64,
                                               float* __restrict__ path) {
    __shared__ float sh[64 * HSTR];
    __shared__ float sg[C_], sb[C_];
    __shared__ float smean[64], srstd[64];
    __shared__ float sproj[4][16][4];
    __shared__ float swt[64][8];     // 4 tap weights (x-sample), 4 tap weights (rpe)
    __shared__ int   sid[64][8];     // x0c,x1c,y0c,y1c (x-sample) then same for rpe

    const int tid  = threadIdx.x;
    const int lane = tid & 31;
    const int wv   = tid >> 5;
    const int b    = blockIdx.x >> 6;    // 64 row-blocks per batch
    const int row  = blockIdx.x & 63;
    const int pixBase = row * 64;
    float* hb = h + (size_t)b * C_ * N_;

    // ---- load h tile (coalesced) + LN params ----
    for (int i = tid; i < 64 * C_; i += 128) {
        int c = i >> 6, p = i & 63;
        sh[p * HSTR + c] = hb[(size_t)c * N_ + pixBase + p];
    }
    for (int i = tid; i < C_; i += 128) { sg[i] = ln_g[i]; sb[i] = ln_b[i]; }

    // ---- B fragments for WMMA (independent of LDS): conv2_w^T padded to 16 cols ----
    v16h bfrag[6];
    {
        const int n  = lane & 15;
        const int kb = (lane < 16) ? 0 : 8;
        for (int t = 0; t < 6; ++t) {
            const int K0 = t * 32;
            v16h bf;
            #pragma unroll
            for (int v = 0; v < 4; ++v) {
                int k  = K0 + kb + 2 * v;
                int k2 = K0 + kb + 16 + 2 * v;
                bf[2 * v]     = (n < 3) ? (_Float16)w2[n * C_ + k]      : (_Float16)0.f;
                bf[2 * v + 1] = (n < 3) ? (_Float16)w2[n * C_ + k + 1]  : (_Float16)0.f;
                bf[8 + 2 * v] = (n < 3) ? (_Float16)w2[n * C_ + k2]     : (_Float16)0.f;
                bf[9 + 2 * v] = (n < 3) ? (_Float16)w2[n * C_ + k2 + 1] : (_Float16)0.f;
            }
            bfrag[t] = bf;
        }
    }
    __syncthreads();

    // ---- per-pixel LayerNorm stats ----
    if (tid < 64) {
        float s = 0.f, s2 = 0.f;
        for (int c = 0; c < C_; ++c) {
            float v = sh[tid * HSTR + c];
            s += v; s2 += v * v;
        }
        float mu  = s * (1.0f / (float)C_);
        float var = s2 * (1.0f / (float)C_) - mu * mu;
        smean[tid] = mu;
        srstd[tid] = rsqrtf(var + LN_EPS);
    }
    __syncthreads();

    // ---- normalize + gelu in place ----
    for (int i = tid; i < 64 * C_; i += 128) {
        int c = i >> 6, p = i & 63;
        float v = (sh[p * HSTR + c] - smean[p]) * srstd[p] * sg[c] + sb[c];
        sh[p * HSTR + c] = gelu_exact(v);
    }
    __syncthreads();

    // ---- WMMA: 16 pixels per wave, K=192 in 6 chunks of 32 ----
    {
        v8f acc = {};
        const int m    = lane & 15;
        const int prow = (wv << 4) + m;
        const int kb   = (lane < 16) ? 0 : 8;
        for (int t = 0; t < 6; ++t) {
            const int K0 = t * 32;
            v16h af;
            #pragma unroll
            for (int v = 0; v < 4; ++v) {
                int k  = K0 + kb + 2 * v;
                int k2 = K0 + kb + 16 + 2 * v;
                af[2 * v]     = (_Float16)sh[prow * HSTR + k];
                af[2 * v + 1] = (_Float16)sh[prow * HSTR + k + 1];
                af[8 + 2 * v] = (_Float16)sh[prow * HSTR + k2];
                af[9 + 2 * v] = (_Float16)sh[prow * HSTR + k2 + 1];
            }
            acc = __builtin_amdgcn_wmma_f32_16x16x32_f16(
                false, af, false, bfrag[t], (short)0, acc, false, false);
        }
        const int n = lane & 15;
        if (n < 3) {
            const int mbase = (lane >= 16) ? 8 : 0;
            #pragma unroll
            for (int r = 0; r < 8; ++r) sproj[wv][mbase + r][n] = acc[r];
        }
    }
    __syncthreads();

    // ---- per-pixel: pos / path_pos / precompute bilinear taps ----
    if (tid < 64) {
        const int p = tid, i = row, j = p;
        float o0 = sproj[p >> 4][p & 15][0];
        float o1 = sproj[p >> 4][p & 15][1];
        float de = sproj[p >> 4][p & 15][2];
        float posy = tanhf(o0) * (1.0f / 63.0f) + ((i + 0.5f) * (2.0f / 63.0f) - 1.0f);
        float posx = tanhf(o1) * (1.0f / 63.0f) + ((j + 0.5f) * (2.0f / 63.0f) - 1.0f);
        path[b * N_ + pixBase + p] =
            tanhf(de) + ((pixBase + p + 0.5f) * (2.0f / 4095.0f) - 1.0f);

        // grid-sample taps on x (align_corners=True, zeros padding)
        float gx = (posx + 1.0f) * 31.5f, gy = (posy + 1.0f) * 31.5f;
        float fx = floorf(gx), fy = floorf(gy);
        float wx = gx - fx, wy = gy - fy;
        int x0 = (int)fx, y0 = (int)fy;
        float vx0 = (x0 >= 0 && x0 < W_) ? 1.f : 0.f;
        float vx1 = (x0 + 1 >= 0 && x0 + 1 < W_) ? 1.f : 0.f;
        float vy0 = (y0 >= 0 && y0 < H_) ? 1.f : 0.f;
        float vy1 = (y0 + 1 >= 0 && y0 + 1 < H_) ? 1.f : 0.f;
        swt[p][0] = (1.f - wx) * (1.f - wy) * vx0 * vy0;
        swt[p][1] = wx * (1.f - wy) * vx1 * vy0;
        swt[p][2] = (1.f - wx) * wy * vx0 * vy1;
        swt[p][3] = wx * wy * vx1 * vy1;
        sid[p][0] = min(max(x0, 0), W_ - 1);
        sid[p][1] = min(max(x0 + 1, 0), W_ - 1);
        sid[p][2] = min(max(y0, 0), H_ - 1);
        sid[p][3] = min(max(y0 + 1, 0), H_ - 1);

        // grid-sample taps on resized rpe at displacement
        float keyy = i * (128.0f / 3969.0f) - 1.0f;
        float keyx = j * (128.0f / 3969.0f) - 1.0f;
        float dx = (keyx - posx) * 0.5f, dy = (keyy - posy) * 0.5f;
        gx = (dx + 1.0f) * 31.5f; gy = (dy + 1.0f) * 31.5f;
        fx = floorf(gx); fy = floorf(gy);
        wx = gx - fx; wy = gy - fy;
        x0 = (int)fx; y0 = (int)fy;
        vx0 = (x0 >= 0 && x0 < W_) ? 1.f : 0.f;
        vx1 = (x0 + 1 >= 0 && x0 + 1 < W_) ? 1.f : 0.f;
        vy0 = (y0 >= 0 && y0 < H_) ? 1.f : 0.f;
        vy1 = (y0 + 1 >= 0 && y0 + 1 < H_) ? 1.f : 0.f;
        swt[p][4] = (1.f - wx) * (1.f - wy) * vx0 * vy0;
        swt[p][5] = wx * (1.f - wy) * vx1 * vy0;
        swt[p][6] = (1.f - wx) * wy * vx0 * vy1;
        swt[p][7] = wx * wy * vx1 * vy1;
        sid[p][4] = min(max(x0, 0), W_ - 1);
        sid[p][5] = min(max(x0 + 1, 0), W_ - 1);
        sid[p][6] = min(max(y0, 0), H_ - 1);
        sid[p][7] = min(max(y0 + 1, 0), H_ - 1);
    }
    __syncthreads();

    // ---- sample all channels, write xo in place over h (coalesced stores) ----
    for (int i = tid; i < 64 * C_; i += 128) {
        int c = i >> 6, p = i & 63;
        const float* xc_ = x + ((size_t)(b * C_ + c)) * N_;
        const float* rc  = rpe64 + (size_t)c * N_;
        float v = swt[p][0] * xc_[sid[p][2] * W_ + sid[p][0]]
                + swt[p][1] * xc_[sid[p][2] * W_ + sid[p][1]]
                + swt[p][2] * xc_[sid[p][3] * W_ + sid[p][0]]
                + swt[p][3] * xc_[sid[p][3] * W_ + sid[p][1]];
        v      += swt[p][4] * rc[sid[p][6] * W_ + sid[p][4]]
                + swt[p][5] * rc[sid[p][6] * W_ + sid[p][5]]
                + swt[p][6] * rc[sid[p][7] * W_ + sid[p][4]]
                + swt[p][7] * rc[sid[p][7] * W_ + sid[p][5]];
        hb[(size_t)c * N_ + pixBase + p] = v;
    }
}

// ---------------- Kernel 6: per-batch bitonic argsort of path_pos (ascending) ----------
__global__ __launch_bounds__(1024) void k_sort(const float* __restrict__ path,
                                               int* __restrict__ idxOut) {
    __shared__ float key[N_];
    __shared__ int   val[N_];
    const int b = blockIdx.x;
    for (int i = threadIdx.x; i < N_; i += 1024) {
        key[i] = path[b * N_ + i];
        val[i] = i;
    }
    __syncthreads();
    for (int k = 2; k <= N_; k <<= 1) {
        for (int j = k >> 1; j > 0; j >>= 1) {
            for (int i = threadIdx.x; i < N_; i += 1024) {
                int ixj = i ^ j;
                if (ixj > i) {
                    bool up = ((i & k) == 0);
                    float ki = key[i], kj = key[ixj];
                    if ((ki > kj) == up) {
                        key[i] = kj; key[ixj] = ki;
                        int t = val[i]; val[i] = val[ixj]; val[ixj] = t;
                    }
                }
            }
            __syncthreads();
        }
    }
    for (int i = threadIdx.x; i < N_; i += 1024) idxOut[b * N_ + i] = val[i];
}

// ---------------- Kernel 7: gather + transpose: out[b,r,c] = xo[b,c,idx[b,r]] ----------
__global__ __launch_bounds__(C_) void k_gather(const float* __restrict__ xo,
                                               const int* __restrict__ idx,
                                               float* __restrict__ out) {
    const int br = blockIdx.x;         // b*N + r
    const int b  = br >> 12;
    const int n  = idx[br];            // uniform -> scalar load
    out[(size_t)br * C_ + threadIdx.x] =
        xo[((size_t)(b * C_) + threadIdx.x) * N_ + n];
}

extern "C" void kernel_launch(void* const* d_in, const int* in_sizes, int n_in,
                              void* d_out, int out_size, void* d_ws, size_t ws_size,
                              hipStream_t stream) {
    (void)in_sizes; (void)n_in; (void)out_size; (void)ws_size;
    const float* x    = (const float*)d_in[0];
    const float* c1w  = (const float*)d_in[1];
    const float* c1b  = (const float*)d_in[2];
    const float* caw1 = (const float*)d_in[3];
    const float* cab1 = (const float*)d_in[4];
    const float* caw2 = (const float*)d_in[5];
    const float* cab2 = (const float*)d_in[6];
    const float* lng  = (const float*)d_in[7];
    const float* lnb  = (const float*)d_in[8];
    const float* c2w  = (const float*)d_in[9];
    const float* rpet = (const float*)d_in[10];

    char* ws = (char*)d_ws;
    float* h     = (float*)ws;                                    // B*C*N (reused as xo)
    float* rpe64 = (float*)(ws + (size_t)B_ * C_ * N_ * 4);       // C*N
    float* xc    = (float*)(ws + ((size_t)B_ * C_ * N_ + (size_t)C_ * N_) * 4);
    float* a     = xc + B_ * C_;
    float* path  = a + B_ * C_;                                   // B*N

    float* xout = (float*)d_out;                                  // (B,N,C) floats
    int* idxOut = (int*)(xout + (size_t)B_ * N_ * C_);            // (B,N) int32

    k_mean  <<<B_ * C_,        256,  0, stream>>>(x, xc);
    k_attn  <<<B_,             C_,   0, stream>>>(xc, caw1, cab1, caw2, cab2, a);
    k_rpe   <<<C_,             256,  0, stream>>>(rpet, rpe64);
    k_conv  <<<B_ * C_ * 16,   256,  0, stream>>>(x, c1w, c1b, a, h);
    k_fused <<<B_ * (N_ / 64), 128,  0, stream>>>(x, h, lng, lnb, c2w, rpe64, path);
    k_sort  <<<B_,             1024, 0, stream>>>(path, idxOut);
    k_gather<<<B_ * N_,        C_,   0, stream>>>(h, idxOut, xout);
}
